// SimpleVectorizer_57054345560160
// MI455X (gfx1250) — matrix operand order repaired and checked by reading, not compiled, not run on hardware
//
#include <hip/hip_runtime.h>

typedef __attribute__((ext_vector_type(8)))  float   v8f;
typedef __attribute__((ext_vector_type(16))) __bf16  v16bf;
typedef __attribute__((ext_vector_type(8)))  __bf16  v8bf;

#define CODEBOOK 1024
#define DIM      256
#define HW       1024            // 32*32
#define BATCH    64
#define NTOK     (BATCH * HW)    // 65536 tokens
#define TILE_M   256             // tokens per workgroup (32 per wave, dual panel)
#define NBLOCKS  (NTOK / TILE_M) // 256
#define AROW     264             // padded LDS row stride (halves): 528B, 16B-aligned
#define KGROUP   128             // codes staged in LDS per group

// d_out layout: [zq | loss | commitment | codebook | min_idx]
#define ZQ_N     (BATCH * DIM * HW)   // 16777216
#define LOSS_OFF ZQ_N
#define IDX_OFF  (ZQ_N + 3)

union BF16Frag { v16bf v; struct { v8bf lo, hi; } p; };

static __device__ __forceinline__ unsigned short f2bf(float f) {
    unsigned int u = __float_as_uint(f);
    unsigned int r = u + 0x7FFFu + ((u >> 16) & 1u);   // round-to-nearest-even
    return (unsigned short)(r >> 16);
}

// ---------------------------------------------------------------------------
// Prep: fp32 codebook -> bf16 copy + ||e||^2 per code; zero loss accumulator.
// ---------------------------------------------------------------------------
__global__ void vq_prep(const float* __restrict__ emb,
                        unsigned short* __restrict__ ebf,
                        float* __restrict__ eNorm,
                        float* __restrict__ accum) {
    __shared__ float sRed[8];
    const int k = blockIdx.x;          // code
    const int c = threadIdx.x;         // channel
    float v = emb[k * DIM + c];
    ebf[k * DIM + c] = f2bf(v);
    float s = v * v;
    #pragma unroll
    for (int off = 16; off > 0; off >>= 1) s += __shfl_xor(s, off, 32);
    const int lane = threadIdx.x & 31, wave = threadIdx.x >> 5;
    if (lane == 0) sRed[wave] = s;
    __syncthreads();
    if (threadIdx.x == 0) {
        float t = 0.f;
        #pragma unroll
        for (int w = 0; w < 8; ++w) t += sRed[w];
        eNorm[k] = t;
        if (k == 0) *accum = 0.f;      // re-zeroed every call (graph-replay safe)
    }
}

// ---------------------------------------------------------------------------
// Main: WMMA distance search + argmin + gather + loss partials.
// Each wave: M=32 tokens (two 16-row A panels sharing every B fragment).
// ---------------------------------------------------------------------------
__global__ void __launch_bounds__(256) vq_main(const float* __restrict__ z,
                                               const float* __restrict__ emb,
                                               const unsigned short* __restrict__ ebf,
                                               const float* __restrict__ eNorm,
                                               float* __restrict__ out,
                                               float* __restrict__ accum) {
    extern __shared__ char smem[];
    unsigned short* sA   = (unsigned short*)smem;                         // 256 x 264 halves
    unsigned short* sB   = (unsigned short*)(smem + TILE_M * AROW * 2);   // 128 x 264 halves
    int*            sIdx = (int*)(smem + TILE_M * AROW * 2 + KGROUP * AROW * 2);
    float*          sRed = (float*)((char*)sIdx + TILE_M * 4);

    const int tid    = threadIdx.x;
    const int b      = blockIdx.x >> 2;            // 4 tiles of 256 tokens per batch
    const int tokOff = (blockIdx.x & 3) * TILE_M;
    const float* zb  = z + (size_t)b * DIM * HW;

    // ---- Phase 1: stage z tile to LDS as bf16 (coalesced per-channel rows)
    for (int i = tid; i < TILE_M * DIM; i += 256) {
        const int m = i & (TILE_M - 1);
        const int c = i >> 8;
        sA[m * AROW + c] = f2bf(zb[(size_t)c * HW + tokOff + m]);
    }
    __syncthreads();

    const int lane  = tid & 31;
    const int wave  = tid >> 5;
    const int half  = lane >> 4;   // lane group (K sub-range selector)
    const int mLane = lane & 15;   // M for A, N for B/C

    // ---- Phase 2: two A panels resident in VGPRs (16x32 bf16 per dc chunk)
    BF16Frag aF0[8], aF1[8];
    const int mRow0 = wave * 32 + mLane;
    const int mRow1 = mRow0 + 16;
    #pragma unroll
    for (int dc = 0; dc < 8; ++dc) {
        const unsigned short* p0 = sA + mRow0 * AROW + dc * 32 + half * 8;
        const unsigned short* p1 = sA + mRow1 * AROW + dc * 32 + half * 8;
        aF0[dc].p.lo = *(const v8bf*)(p0);
        aF0[dc].p.hi = *(const v8bf*)(p0 + 16);
        aF1[dc].p.lo = *(const v8bf*)(p1);
        aF1[dc].p.hi = *(const v8bf*)(p1 + 16);
    }

    float bestVal0[8], bestVal1[8];
    int   bestIdx0[8], bestIdx1[8];
    #pragma unroll
    for (int r = 0; r < 8; ++r) {
        bestVal0[r] = 3.4e38f; bestIdx0[r] = 0;
        bestVal1[r] = 3.4e38f; bestIdx1[r] = 0;
    }

    const unsigned int* ebf32 = (const unsigned int*)ebf;

    auto loadB = [&](BF16Frag& f, int chunk, int dc) {
        const unsigned short* bb = sB + (chunk * 16 + mLane) * AROW + dc * 32 + half * 8;
        f.p.lo = *(const v8bf*)(bb);
        f.p.hi = *(const v8bf*)(bb + 16);
    };

    // ---- Phase 3: loop code groups; B staged in LDS, shared by all 8 waves
    #pragma unroll 1
    for (int g = 0; g < CODEBOOK / KGROUP; ++g) {
        __syncthreads();   // previous-group B reads complete
        for (int i = tid; i < KGROUP * (DIM / 2); i += 256) {
            const int kk = i >> 7;       // local code 0..127
            const int c2 = i & 127;      // bf16-pair index
            ((unsigned int*)(sB + kk * AROW))[c2] =
                ebf32[(size_t)(g * KGROUP + kk) * (DIM / 2) + c2];
        }
        __syncthreads();

        // per-group ||e||^2 preload (latency hidden under first chunks)
        float en8[8];
        #pragma unroll
        for (int ch = 0; ch < 8; ++ch) en8[ch] = eNorm[g * KGROUP + ch * 16 + mLane];

        // software-pipelined B fragments: stage s+1 in flight during stage s WMMAs
        BF16Frag bPing, bPong;
        loadB(bPing, 0, 0);
        #pragma unroll
        for (int chunk = 0; chunk < KGROUP / 16; ++chunk) {
            v8f acc0 = {}, acc1 = {};
            #pragma unroll
            for (int dc = 0; dc < 8; ++dc) {
                const int s = chunk * 8 + dc;
                BF16Frag& cur = (s & 1) ? bPong : bPing;
                BF16Frag& nxt = (s & 1) ? bPing : bPong;
                if (s < 63) loadB(nxt, (dc == 7) ? chunk + 1 : chunk, (dc + 1) & 7);
                acc0 = __builtin_amdgcn_wmma_f32_16x16x32_bf16(
                         false, aF0[dc].v, false, cur.v, (short)0, acc0, false, false);
                acc1 = __builtin_amdgcn_wmma_f32_16x16x32_bf16(
                         false, aF1[dc].v, false, cur.v, (short)0, acc1, false, false);
            }
            // distance surrogate: ||e||^2 - 2 z.e   (||z||^2 constant per token)
            const int   kGlob = g * KGROUP + chunk * 16 + mLane;
            const float en    = en8[chunk];
            #pragma unroll
            for (int r = 0; r < 8; ++r) {            // C: VGPR r -> M = r + 8*half
                const float d0 = __builtin_fmaf(-2.f, acc0[r], en);
                const float d1 = __builtin_fmaf(-2.f, acc1[r], en);
                if (d0 < bestVal0[r]) { bestVal0[r] = d0; bestIdx0[r] = kGlob; }
                if (d1 < bestVal1[r]) { bestVal1[r] = d1; bestIdx1[r] = kGlob; }
            }
        }
    }

    // ---- Phase 4: argmin across the 16 N-lanes of each half-group
    #pragma unroll
    for (int r = 0; r < 8; ++r) {
        float v0 = bestVal0[r], v1 = bestVal1[r];
        int   i0 = bestIdx0[r], i1 = bestIdx1[r];
        #pragma unroll
        for (int off = 1; off < 16; off <<= 1) {
            const float o0 = __shfl_xor(v0, off, 32);
            const int   j0 = __shfl_xor(i0, off, 32);
            const float o1 = __shfl_xor(v1, off, 32);
            const int   j1 = __shfl_xor(i1, off, 32);
            if (o0 < v0 || (o0 == v0 && j0 < i0)) { v0 = o0; i0 = j0; }
            if (o1 < v1 || (o1 == v1 && j1 < i1)) { v1 = o1; i1 = j1; }
        }
        if (mLane == 0) {
            sIdx[wave * 32 + half * 8 + r]      = i0;   // rows 0..15 of panel
            sIdx[wave * 32 + 16 + half * 8 + r] = i1;   // rows 16..31
        }
    }
    __syncthreads();

    // ---- Phase 5: gather zq (fp32 codebook), write BCHW, loss partial
    float lsum = 0.f;
    const size_t outBase = (size_t)b * DIM * HW + tokOff;
    for (int i = tid; i < TILE_M * DIM; i += 256) {
        const int m   = i & (TILE_M - 1);
        const int c   = i >> 8;
        const int idx = sIdx[m];
        const float ev = emb[(size_t)idx * DIM + c];
        const float zv = zb[(size_t)c * HW + tokOff + m];
        const float d  = ev - zv;
        lsum += d * d;
        out[outBase + (size_t)c * HW + m] = ev;
    }
    #pragma unroll
    for (int off = 16; off > 0; off >>= 1) lsum += __shfl_xor(lsum, off, 32);
    if (lane == 0) sRed[wave] = lsum;
    __syncthreads();
    if (tid == 0) {
        float s = 0.f;
        #pragma unroll
        for (int w = 0; w < 8; ++w) s += sRed[w];
        atomicAdd(accum, s);
    }
    {   // min_idx output (as float), one token per thread
        const int n = b * HW + tokOff + tid;
        out[IDX_OFF + n] = (float)sIdx[tid];
    }
}

// ---------------------------------------------------------------------------
// Epilogue: finalize the three loss scalars.
// ---------------------------------------------------------------------------
__global__ void vq_final(const float* __restrict__ accum, float* __restrict__ out) {
    const float mse = *accum * (1.0f / (float)ZQ_N);
    out[LOSS_OFF + 0] = 1.25f * mse;   // loss = commitment + codebook
    out[LOSS_OFF + 1] = 0.25f * mse;   // commitment
    out[LOSS_OFF + 2] = mse;           // codebook
}

extern "C" void kernel_launch(void* const* d_in, const int* in_sizes, int n_in,
                              void* d_out, int out_size, void* d_ws, size_t ws_size,
                              hipStream_t stream) {
    const float* z   = (const float*)d_in[0];
    const float* emb = (const float*)d_in[1];
    float* out = (float*)d_out;

    char* ws = (char*)d_ws;
    float*          accum = (float*)ws;                                     // 4 B
    unsigned short* ebf   = (unsigned short*)(ws + 64);                     // 512 KB
    float*          eNorm = (float*)(ws + 64 + CODEBOOK * DIM * 2);         // 4 KB

    vq_prep<<<CODEBOOK, 256, 0, stream>>>(emb, ebf, eNorm, accum);

    const size_t lds = (TILE_M + KGROUP) * AROW * 2 + TILE_M * 4 + 64;
    vq_main<<<NBLOCKS, 256, lds, stream>>>(z, emb, ebf, eNorm, out, accum);

    vq_final<<<1, 1, 0, stream>>>(accum, out);
}
